// Model_34883724378143
// MI455X (gfx1250) — compile-verified, matrix operand order checked by reading, not run on hardware
//
#include <hip/hip_runtime.h>
#include <hip/hip_bf16.h>

typedef __attribute__((ext_vector_type(16))) _Float16 v16h;
typedef __attribute__((ext_vector_type(8)))  _Float16 v8h;
typedef __attribute__((ext_vector_type(8)))  float    v8f;

#define B_    16
#define S_    512
#define C_    7
#define D_    512
#define H_    8
#define E_    64
#define DFF_  2048
#define L_    3
#define NCLS_ 10
#define PH_   64

// ---------------------------------------------------------------- helpers

__device__ __forceinline__ v8h cvt8(float4 a, float4 b) {
  v8h r;
  r[0] = (_Float16)a.x; r[1] = (_Float16)a.y; r[2] = (_Float16)a.z; r[3] = (_Float16)a.w;
  r[4] = (_Float16)b.x; r[5] = (_Float16)b.y; r[6] = (_Float16)b.z; r[7] = (_Float16)b.w;
  return r;
}

__device__ __forceinline__ v16h mk16(v8h lo, v8h hi) {
  v16h r;
#pragma unroll
  for (int i = 0; i < 8; ++i) { r[i] = lo[i]; r[i + 8] = hi[i]; }
  return r;
}

__device__ __forceinline__ v8f wmma_f16(v16h a, v16h b, v8f c) {
  return __builtin_amdgcn_wmma_f32_16x16x32_f16(false, a, false, b, (short)0, c, false, false);
}

__device__ __forceinline__ float gelu_exact(float x) {
  return 0.5f * x * (1.0f + erff(x * 0.7071067811865475f));
}

// ---------------------------------------------------------------- small kernels

// mean/std over S for each (b,c):  x_enc (B,S,C)
__global__ __launch_bounds__(128) void stats_kernel(const float* __restrict__ x,
                                                    float* __restrict__ meanc,
                                                    float* __restrict__ stdc) {
  __shared__ float rs[128], rq[128];
  int bc = blockIdx.x;               // 0..B*C-1
  int b = bc / C_, c = bc % C_;
  int tid = threadIdx.x;
  float s = 0.f, q = 0.f;
  for (int i = tid; i < S_; i += 128) {
    float v = x[(size_t)b * S_ * C_ + (size_t)i * C_ + c];
    s += v; q += v * v;
  }
  rs[tid] = s; rq[tid] = q; __syncthreads();
  for (int o = 64; o > 0; o >>= 1) {
    if (tid < o) { rs[tid] += rs[tid + o]; rq[tid] += rq[tid + o]; }
    __syncthreads();
  }
  if (tid == 0) {
    float mu = rs[0] / (float)S_;
    float var = rq[0] / (float)S_ - mu * mu;
    meanc[bc] = mu;
    stdc[bc]  = sqrtf(var + 1e-5f);
  }
}

// ycat[b, j], j<C: circular conv over C axis (in-ch S), j>=C: stats
__global__ __launch_bounds__(128) void projconv_kernel(const float* __restrict__ x,
                                                       const float* __restrict__ cw,   // (1,S,3)
                                                       const float* __restrict__ stat, // (B,C)
                                                       float* __restrict__ ycat) {     // (B,2C)
  __shared__ float red[128];
  int bj = blockIdx.x;
  int b = bj / (2 * C_), j = bj % (2 * C_);
  int tid = threadIdx.x;
  if (j >= C_) {
    if (tid == 0) ycat[(size_t)b * 2 * C_ + j] = stat[(size_t)b * C_ + (j - C_)];
    return;
  }
  float s = 0.f;
  for (int i = tid; i < S_; i += 128) {
    const float* xr = x + (size_t)b * S_ * C_ + (size_t)i * C_;
    const float* wr = cw + (size_t)i * 3;
#pragma unroll
    for (int k = 0; k < 3; ++k) {
      int c = (j + k - 1 + C_) % C_;
      s += xr[c] * wr[k];
    }
  }
  red[tid] = s; __syncthreads();
  for (int o = 64; o > 0; o >>= 1) { if (tid < o) red[tid] += red[tid + o]; __syncthreads(); }
  if (tid == 0) ycat[(size_t)b * 2 * C_ + j] = red[0];
}

// h = relu(ycat @ w1.T + b1)
__global__ __launch_bounds__(128) void projmlp_kernel(const float* __restrict__ ycat,
                                                      const float* __restrict__ w1, // (PH,2C)
                                                      const float* __restrict__ b1,
                                                      float* __restrict__ h) {      // (B,PH)
  int idx = blockIdx.x * 128 + threadIdx.x;
  if (idx >= B_ * PH_) return;
  int b = idx / PH_, p = idx % PH_;
  float s = b1[p];
#pragma unroll
  for (int j = 0; j < 2 * C_; ++j)
    s += ycat[(size_t)b * 2 * C_ + j] * w1[(size_t)p * 2 * C_ + j];
  h[idx] = fmaxf(s, 0.f);
}

__global__ __launch_bounds__(32) void tau_kernel(const float* __restrict__ h,
                                                 const float* __restrict__ w2, // (1,PH)
                                                 float* __restrict__ tau) {
  int b = threadIdx.x;
  if (b >= B_) return;
  float s = 0.f;
#pragma unroll
  for (int p = 0; p < PH_; ++p) s += h[(size_t)b * PH_ + p] * w2[p];
  tau[b] = expf(s);
}

__global__ __launch_bounds__(128) void delta_kernel(const float* __restrict__ h,
                                                    const float* __restrict__ w2, // (S,PH)
                                                    float* __restrict__ delta) {  // (B,S)
  int idx = blockIdx.x * 128 + threadIdx.x;
  if (idx >= B_ * S_) return;
  int b = idx / S_, s = idx % S_;
  float acc = 0.f;
#pragma unroll 8
  for (int p = 0; p < PH_; ++p) acc += h[(size_t)b * PH_ + p] * w2[(size_t)s * PH_ + p];
  delta[idx] = acc;
}

// X[b,s,d] = sum_{c,k} x_enc[b,(s+k-1)%S,c]*tok_w[d,c,k] + pe[s,d]
__global__ __launch_bounds__(256) void embed_kernel(const float* __restrict__ x,
                                                    const float* __restrict__ tw, // (D,C,3)
                                                    float* __restrict__ X) {
  size_t idx = (size_t)blockIdx.x * 256 + threadIdx.x;
  if (idx >= (size_t)B_ * S_ * D_) return;
  int d = idx % D_;
  int s = (idx / D_) % S_;
  int b = idx / ((size_t)S_ * D_);
  float acc = 0.f;
#pragma unroll
  for (int k = 0; k < 3; ++k) {
    int sp = s + k - 1; sp = (sp + S_) % S_;
    const float* xr = x + (size_t)b * S_ * C_ + (size_t)sp * C_;
    const float* wr = tw + (size_t)d * C_ * 3 + k;
#pragma unroll
    for (int c = 0; c < C_; ++c) acc += xr[c] * wr[c * 3];
  }
  int j = d >> 1;
  float div = __expf(-(float)(2 * j) * (9.210340371976184f / (float)D_)); // ln(10000)
  float pe = (d & 1) ? __cosf((float)s * div) : __sinf((float)s * div);
  X[idx] = acc + pe;
}

// ---------------------------------------------------------------- WMMA GEMM
// C[M,N] = act(A[M,K] @ W[N,K]^T + bias[N])
// REQUIRES: M % 128 == 0, N % 128 == 0, K % 32 == 0 (true for all uses here).
// Block tile 128x128, 8 waves (4x2), wave tile 32x64, double-buffered LDS,
// software-pipelined global loads + prefetch of tile k+2.
__global__ __launch_bounds__(256) void gemm_wmma_kernel(const float* __restrict__ A,
                                                        const float* __restrict__ W,
                                                        const float* __restrict__ bias,
                                                        float* __restrict__ Cout,
                                                        int M, int N, int K, int act) {
  __shared__ __align__(16) _Float16 sA[2][128][40];
  __shared__ __align__(16) _Float16 sB[2][128][40];

  const int tid  = threadIdx.x;
  const int lane = tid & 31;
  const int wave = tid >> 5;
  const int wm = wave >> 1;           // 0..3 (32-row slice)
  const int wn = wave & 1;            // 0..1 (64-col slice)
  const int bm = blockIdx.x * 128;
  const int bn = blockIdx.y * 128;
  const int krun = 8 * (lane >> 4);   // 0 or 8
  const int l16  = lane & 15;

  // staging map: 2 threads per row, 16 halves each
  const int sRow = tid >> 1;
  const int sCol = (tid & 1) * 16;

  const float* aptr = A + (size_t)(bm + sRow) * K + sCol;
  const float* bptr = W + (size_t)(bn + sRow) * K + sCol;

  v8f acc[2][4];
#pragma unroll
  for (int mt = 0; mt < 2; ++mt)
#pragma unroll
    for (int nt = 0; nt < 4; ++nt) acc[mt][nt] = (v8f){0,0,0,0,0,0,0,0};

  const int nk = K >> 5;

  // prologue: stage tile 0 into buffer 0
  {
    float4 fA[4], fB[4];
#pragma unroll
    for (int j = 0; j < 4; ++j) {
      fA[j] = ((const float4*)aptr)[j];
      fB[j] = ((const float4*)bptr)[j];
    }
    *(v8h*)&sA[0][sRow][sCol]     = cvt8(fA[0], fA[1]);
    *(v8h*)&sA[0][sRow][sCol + 8] = cvt8(fA[2], fA[3]);
    *(v8h*)&sB[0][sRow][sCol]     = cvt8(fB[0], fB[1]);
    *(v8h*)&sB[0][sRow][sCol + 8] = cvt8(fB[2], fB[3]);
  }

  for (int kt = 0; kt < nk; ++kt) {
    const int cur = kt & 1;
    const int nxt = cur ^ 1;
    const bool hasNext = (kt + 1 < nk);

    float4 fA[4], fB[4];
    if (hasNext) {
      const float* ap = aptr + (size_t)(kt + 1) * 32;
      const float* bp = bptr + (size_t)(kt + 1) * 32;
#pragma unroll
      for (int j = 0; j < 4; ++j) {
        fA[j] = ((const float4*)ap)[j];
        fB[j] = ((const float4*)bp)[j];
      }
      if (kt + 2 < nk) {
        __builtin_prefetch(aptr + (size_t)(kt + 2) * 32, 0, 1);
        __builtin_prefetch(bptr + (size_t)(kt + 2) * 32, 0, 1);
      }
    }

    __syncthreads();  // stores of `cur` tile are visible

    v16h am[2], bf[4];
#pragma unroll
    for (int mt = 0; mt < 2; ++mt) {
      int row = wm * 32 + mt * 16 + l16;
      v8h lo = *(const v8h*)&sA[cur][row][krun];
      v8h hi = *(const v8h*)&sA[cur][row][krun + 16];
      am[mt] = mk16(lo, hi);
    }
#pragma unroll
    for (int nt = 0; nt < 4; ++nt) {
      int row = wn * 64 + nt * 16 + l16;
      v8h lo = *(const v8h*)&sB[cur][row][krun];
      v8h hi = *(const v8h*)&sB[cur][row][krun + 16];
      bf[nt] = mk16(lo, hi);
    }
#pragma unroll
    for (int mt = 0; mt < 2; ++mt)
#pragma unroll
      for (int nt = 0; nt < 4; ++nt)
        acc[mt][nt] = wmma_f16(am[mt], bf[nt], acc[mt][nt]);

    if (hasNext) {
      *(v8h*)&sA[nxt][sRow][sCol]     = cvt8(fA[0], fA[1]);
      *(v8h*)&sA[nxt][sRow][sCol + 8] = cvt8(fA[2], fA[3]);
      *(v8h*)&sB[nxt][sRow][sCol]     = cvt8(fB[0], fB[1]);
      *(v8h*)&sB[nxt][sRow][sCol + 8] = cvt8(fB[2], fB[3]);
    }
  }

  // epilogue (no bounds checks: exact tiling)
#pragma unroll
  for (int nt = 0; nt < 4; ++nt) {
    int gcol = bn + wn * 64 + nt * 16 + l16;
    float bv = bias[gcol];
#pragma unroll
    for (int mt = 0; mt < 2; ++mt) {
#pragma unroll
      for (int r = 0; r < 8; ++r) {
        int grow = bm + wm * 32 + mt * 16 + r + 8 * (lane >> 4);
        float v = acc[mt][nt][r] + bv;
        if (act == 2) v = gelu_exact(v);
        Cout[(size_t)grow * N + gcol] = v;
      }
    }
  }
}

// ---------------------------------------------------------------- attention
// per block: one (b,h) and 64 query rows; flash-style online softmax
__global__ __launch_bounds__(128) void attn_kernel(const float* __restrict__ Qg,
                                                   const float* __restrict__ Kg,
                                                   const float* __restrict__ Vg,
                                                   float* __restrict__ Og,
                                                   const float* __restrict__ tau,
                                                   const float* __restrict__ delta) {
  __shared__ __align__(16) _Float16 sK[64][72];
  __shared__ __align__(16) _Float16 sVT[64][72];
  __shared__ __align__(16) _Float16 sP[4][16][72];

  const int tid  = threadIdx.x;
  const int lane = tid & 31;
  const int wave = tid >> 5;
  const int l16  = lane & 15;
  const int kr   = 8 * (lane >> 4);

  const int tiles = S_ / 64;
  int blk = blockIdx.x;
  int qt = blk % tiles;
  int bh = blk / tiles;
  int h = bh % H_;
  int b = bh / H_;

  const float scale = 0.125f; // 1/sqrt(E)
  const float at = scale * tau[b];

  // Q fragments for this wave's 16 rows (persist across key tiles)
  int qrow = qt * 64 + wave * 16 + l16;
  const float* qp = Qg + ((size_t)(b * S_ + qrow)) * D_ + h * E_;
  v16h aq[2];
#pragma unroll
  for (int kk = 0; kk < 2; ++kk) {
    const float* p0 = qp + kk * 32 + kr;
    float4 f0 = ((const float4*)p0)[0];
    float4 f1 = ((const float4*)p0)[1];
    float4 f2 = ((const float4*)(p0 + 16))[0];
    float4 f3 = ((const float4*)(p0 + 16))[1];
    aq[kk] = mk16(cvt8(f0, f1), cvt8(f2, f3));
  }

  v8f oacc[4];
#pragma unroll
  for (int nt = 0; nt < 4; ++nt) oacc[nt] = (v8f){0,0,0,0,0,0,0,0};
  float mrow[8], lsum[8];
#pragma unroll
  for (int r = 0; r < 8; ++r) { mrow[r] = -1e30f; lsum[r] = 0.f; }

  for (int kt = 0; kt < tiles; ++kt) {
    __syncthreads();
    // stage K (native) and V (transposed) tiles, fp32 -> f16
    {
      int r = tid >> 1;
      int c0 = (tid & 1) * 32;
      const float* kp = Kg + ((size_t)(b * S_ + kt * 64 + r)) * D_ + h * E_ + c0;
      const float* vp = Vg + ((size_t)(b * S_ + kt * 64 + r)) * D_ + h * E_ + c0;
#pragma unroll
      for (int j = 0; j < 4; ++j) {
        float4 f0 = ((const float4*)kp)[2 * j];
        float4 f1 = ((const float4*)kp)[2 * j + 1];
        *(v8h*)&sK[r][c0 + j * 8] = cvt8(f0, f1);
      }
#pragma unroll
      for (int j = 0; j < 8; ++j) {
        float4 f = ((const float4*)vp)[j];
        int e = c0 + j * 4;
        sVT[e + 0][r] = (_Float16)f.x;
        sVT[e + 1][r] = (_Float16)f.y;
        sVT[e + 2][r] = (_Float16)f.z;
        sVT[e + 3][r] = (_Float16)f.w;
      }
    }
    __syncthreads();

    // scores: Q (16x64) x K^T (64x64) -> 4 accumulators of 16x16
    v8f sc[4];
#pragma unroll
    for (int nt = 0; nt < 4; ++nt) sc[nt] = (v8f){0,0,0,0,0,0,0,0};
#pragma unroll
    for (int kk = 0; kk < 2; ++kk) {
#pragma unroll
      for (int nt = 0; nt < 4; ++nt) {
        int n = nt * 16 + l16;
        v8h lo = *(const v8h*)&sK[n][kk * 32 + kr];
        v8h hi = *(const v8h*)&sK[n][kk * 32 + kr + 16];
        sc[nt] = wmma_f16(aq[kk], mk16(lo, hi), sc[nt]);
      }
    }

    float dl[4];
#pragma unroll
    for (int nt = 0; nt < 4; ++nt)
      dl[nt] = scale * delta[(size_t)b * S_ + kt * 64 + nt * 16 + l16];

    // online softmax per row (C-layout: lane = column, VGPR index = row)
#pragma unroll
    for (int r = 0; r < 8; ++r) {
      float l0 = sc[0][r] * at + dl[0];
      float l1 = sc[1][r] * at + dl[1];
      float l2 = sc[2][r] * at + dl[2];
      float l3 = sc[3][r] * at + dl[3];
      float mx = fmaxf(fmaxf(l0, l1), fmaxf(l2, l3));
#pragma unroll
      for (int off = 8; off > 0; off >>= 1) mx = fmaxf(mx, __shfl_xor(mx, off, 32));
      float mnew = fmaxf(mrow[r], mx);
      float corr = __expf(mrow[r] - mnew);
      float p0 = __expf(l0 - mnew);
      float p1 = __expf(l1 - mnew);
      float p2 = __expf(l2 - mnew);
      float p3 = __expf(l3 - mnew);
      float ps = p0 + p1 + p2 + p3;
#pragma unroll
      for (int off = 8; off > 0; off >>= 1) ps += __shfl_xor(ps, off, 32);
      lsum[r] = lsum[r] * corr + ps;
      mrow[r] = mnew;
#pragma unroll
      for (int nt = 0; nt < 4; ++nt) oacc[nt][r] *= corr;
      int prow = r + 8 * (lane >> 4);
      sP[wave][prow][0  + l16] = (_Float16)p0;
      sP[wave][prow][16 + l16] = (_Float16)p1;
      sP[wave][prow][32 + l16] = (_Float16)p2;
      sP[wave][prow][48 + l16] = (_Float16)p3;
    }

    // PV: P (16x64) x V (64x64); per-wave LDS region, same-wave DS ordering
#pragma unroll
    for (int kk = 0; kk < 2; ++kk) {
      v8h plo = *(const v8h*)&sP[wave][l16][kk * 32 + kr];
      v8h phi = *(const v8h*)&sP[wave][l16][kk * 32 + kr + 16];
      v16h ap = mk16(plo, phi);
#pragma unroll
      for (int nt = 0; nt < 4; ++nt) {
        int e = nt * 16 + l16;
        v8h blo = *(const v8h*)&sVT[e][kk * 32 + kr];
        v8h bhi = *(const v8h*)&sVT[e][kk * 32 + kr + 16];
        oacc[nt] = wmma_f16(ap, mk16(blo, bhi), oacc[nt]);
      }
    }
  }

  // write O normalized by row sums
#pragma unroll
  for (int nt = 0; nt < 4; ++nt) {
    int e = h * E_ + nt * 16 + l16;
#pragma unroll
    for (int r = 0; r < 8; ++r) {
      int row = qt * 64 + wave * 16 + r + 8 * (lane >> 4);
      Og[((size_t)(b * S_ + row)) * D_ + e] = oacc[nt][r] / lsum[r];
    }
  }
}

// ---------------------------------------------------------------- layernorm
// x = LN(x + add) * g + b  (in place, one row of 512 per block)
__global__ __launch_bounds__(128) void ln_add_kernel(float* __restrict__ x,
                                                     const float* __restrict__ add,
                                                     const float* __restrict__ g,
                                                     const float* __restrict__ bb) {
  __shared__ float red[128];
  int row = blockIdx.x, tid = threadIdx.x;
  size_t base = (size_t)row * D_;
  float v[4]; float s = 0.f;
#pragma unroll
  for (int j = 0; j < 4; ++j) {
    int c = tid + j * 128;
    v[j] = x[base + c] + add[base + c];
    s += v[j];
  }
  red[tid] = s; __syncthreads();
  for (int o = 64; o > 0; o >>= 1) { if (tid < o) red[tid] += red[tid + o]; __syncthreads(); }
  float mu = red[0] / (float)D_;
  __syncthreads();
  s = 0.f;
#pragma unroll
  for (int j = 0; j < 4; ++j) { float d = v[j] - mu; s += d * d; }
  red[tid] = s; __syncthreads();
  for (int o = 64; o > 0; o >>= 1) { if (tid < o) red[tid] += red[tid + o]; __syncthreads(); }
  float rs = rsqrtf(red[0] / (float)D_ + 1e-5f);
#pragma unroll
  for (int j = 0; j < 4; ++j) {
    int c = tid + j * 128;
    x[base + c] = (v[j] - mu) * rs * g[c] + bb[c];
  }
}

// out = gelu(LN(x)*g+b) * mark[row]
__global__ __launch_bounds__(128) void lnf_gelu_kernel(const float* __restrict__ x,
                                                       const float* __restrict__ g,
                                                       const float* __restrict__ bb,
                                                       const float* __restrict__ mark,
                                                       float* __restrict__ out) {
  __shared__ float red[128];
  int row = blockIdx.x, tid = threadIdx.x;
  size_t base = (size_t)row * D_;
  float mk = mark[row];
  float v[4]; float s = 0.f;
#pragma unroll
  for (int j = 0; j < 4; ++j) { v[j] = x[base + tid + j * 128]; s += v[j]; }
  red[tid] = s; __syncthreads();
  for (int o = 64; o > 0; o >>= 1) { if (tid < o) red[tid] += red[tid + o]; __syncthreads(); }
  float mu = red[0] / (float)D_;
  __syncthreads();
  s = 0.f;
#pragma unroll
  for (int j = 0; j < 4; ++j) { float d = v[j] - mu; s += d * d; }
  red[tid] = s; __syncthreads();
  for (int o = 64; o > 0; o >>= 1) { if (tid < o) red[tid] += red[tid + o]; __syncthreads(); }
  float rs = rsqrtf(red[0] / (float)D_ + 1e-5f);
#pragma unroll
  for (int j = 0; j < 4; ++j) {
    int c = tid + j * 128;
    float y = (v[j] - mu) * rs * g[c] + bb[c];
    out[base + c] = gelu_exact(y) * mk;
  }
}

// out[b,n] = dot(act[b,:], proj_w[n,:]) + proj_b[n];  K = S*D
__global__ __launch_bounds__(256) void head_kernel(const float* __restrict__ act,
                                                   const float* __restrict__ pw,
                                                   const float* __restrict__ pb,
                                                   float* __restrict__ out) {
  __shared__ float red[256];
  int b = blockIdx.x / NCLS_, n = blockIdx.x % NCLS_;
  int tid = threadIdx.x;
  const size_t Kk = (size_t)S_ * D_;
  const float4* a4 = (const float4*)(act + (size_t)b * Kk);
  const float4* w4 = (const float4*)(pw + (size_t)n * Kk);
  float s = 0.f;
  for (size_t k = tid; k < Kk / 4; k += 256) {
    float4 av = a4[k], wv = w4[k];
    s += av.x * wv.x + av.y * wv.y + av.z * wv.z + av.w * wv.w;
  }
  red[tid] = s; __syncthreads();
  for (int o = 128; o > 0; o >>= 1) { if (tid < o) red[tid] += red[tid + o]; __syncthreads(); }
  if (tid == 0) out[b * NCLS_ + n] = red[0] + pb[n];
}

// ---------------------------------------------------------------- launch

extern "C" void kernel_launch(void* const* d_in, const int* in_sizes, int n_in,
                              void* d_out, int out_size, void* d_ws, size_t ws_size,
                              hipStream_t stream) {
  (void)in_sizes; (void)n_in; (void)out_size; (void)ws_size;

  const float* x_enc    = (const float*)d_in[0];
  const float* x_mark   = (const float*)d_in[1];
  const float* tok_w    = (const float*)d_in[2];
  const float* Wq       = (const float*)d_in[3];
  const float* bq       = (const float*)d_in[4];
  const float* Wk       = (const float*)d_in[5];
  const float* bk       = (const float*)d_in[6];
  const float* Wv       = (const float*)d_in[7];
  const float* bv       = (const float*)d_in[8];
  const float* Wo       = (const float*)d_in[9];
  const float* bo       = (const float*)d_in[10];
  const float* W1       = (const float*)d_in[11];
  const float* b1       = (const float*)d_in[12];
  const float* W2       = (const float*)d_in[13];
  const float* b2       = (const float*)d_in[14];
  const float* ln1_g    = (const float*)d_in[15];
  const float* ln1_b    = (const float*)d_in[16];
  const float* ln2_g    = (const float*)d_in[17];
  const float* ln2_b    = (const float*)d_in[18];
  const float* lnf_g    = (const float*)d_in[19];
  const float* lnf_b    = (const float*)d_in[20];
  const float* proj_w   = (const float*)d_in[21];
  const float* proj_b   = (const float*)d_in[22];
  const float* tau_conv = (const float*)d_in[23];
  const float* tau_w1   = (const float*)d_in[24];
  const float* tau_b1   = (const float*)d_in[25];
  const float* tau_w2   = (const float*)d_in[26];
  const float* del_conv = (const float*)d_in[27];
  const float* del_w1   = (const float*)d_in[28];
  const float* del_b1   = (const float*)d_in[29];
  const float* del_w2   = (const float*)d_in[30];

  float* ws = (float*)d_ws;
  const size_t NBSD = (size_t)B_ * S_ * D_;
  float* X   = ws;
  float* Qb  = X + NBSD;
  float* Kb  = Qb + NBSD;
  float* Vb  = Kb + NBSD;
  float* Ob  = Vb + NBSD;
  float* Ab  = Ob + NBSD;
  float* FF  = Ab + NBSD;
  float* sm  = FF + (size_t)B_ * S_ * DFF_;
  float* tau    = sm; sm += B_;
  float* delta  = sm; sm += B_ * S_;
  float* meanc  = sm; sm += B_ * C_;
  float* stdc   = sm; sm += B_ * C_;
  float* ycat_t = sm; sm += B_ * 2 * C_;
  float* ycat_d = sm; sm += B_ * 2 * C_;
  float* h_t    = sm; sm += B_ * PH_;
  float* h_d    = sm; sm += B_ * PH_;

  stats_kernel<<<B_ * C_, 128, 0, stream>>>(x_enc, meanc, stdc);
  projconv_kernel<<<B_ * 2 * C_, 128, 0, stream>>>(x_enc, tau_conv, stdc, ycat_t);
  projconv_kernel<<<B_ * 2 * C_, 128, 0, stream>>>(x_enc, del_conv, meanc, ycat_d);
  projmlp_kernel<<<(B_ * PH_ + 127) / 128, 128, 0, stream>>>(ycat_t, tau_w1, tau_b1, h_t);
  projmlp_kernel<<<(B_ * PH_ + 127) / 128, 128, 0, stream>>>(ycat_d, del_w1, del_b1, h_d);
  tau_kernel<<<1, 32, 0, stream>>>(h_t, tau_w2, tau);
  delta_kernel<<<(B_ * S_ + 127) / 128, 128, 0, stream>>>(h_d, del_w2, delta);
  embed_kernel<<<(int)(((size_t)B_ * S_ * D_ + 255) / 256), 256, 0, stream>>>(x_enc, tok_w, X);

  dim3 gD((B_ * S_) / 128, D_ / 128);     // 64 x 4
  dim3 gF((B_ * S_) / 128, DFF_ / 128);   // 64 x 16
  for (int l = 0; l < L_; ++l) {
    gemm_wmma_kernel<<<gD, 256, 0, stream>>>(X, Wq + (size_t)l * D_ * D_, bq + l * D_, Qb, B_ * S_, D_, D_, 0);
    gemm_wmma_kernel<<<gD, 256, 0, stream>>>(X, Wk + (size_t)l * D_ * D_, bk + l * D_, Kb, B_ * S_, D_, D_, 0);
    gemm_wmma_kernel<<<gD, 256, 0, stream>>>(X, Wv + (size_t)l * D_ * D_, bv + l * D_, Vb, B_ * S_, D_, D_, 0);
    attn_kernel<<<B_ * H_ * (S_ / 64), 128, 0, stream>>>(Qb, Kb, Vb, Ob, tau, delta);
    gemm_wmma_kernel<<<gD, 256, 0, stream>>>(Ob, Wo + (size_t)l * D_ * D_, bo + l * D_, Ab, B_ * S_, D_, D_, 0);
    ln_add_kernel<<<B_ * S_, 128, 0, stream>>>(X, Ab, ln1_g + l * D_, ln1_b + l * D_);
    gemm_wmma_kernel<<<gF, 256, 0, stream>>>(X, W1 + (size_t)l * DFF_ * D_, b1 + l * DFF_, FF, B_ * S_, DFF_, D_, 2);
    gemm_wmma_kernel<<<gD, 256, 0, stream>>>(FF, W2 + (size_t)l * D_ * DFF_, b2 + l * D_, Ab, B_ * S_, D_, DFF_, 0);
    ln_add_kernel<<<B_ * S_, 128, 0, stream>>>(X, Ab, ln2_g + l * D_, ln2_b + l * D_);
  }
  lnf_gelu_kernel<<<B_ * S_, 128, 0, stream>>>(X, lnf_g, lnf_b, x_mark, Qb);
  head_kernel<<<B_ * NCLS_, 256, 0, stream>>>(Qb, proj_w, proj_b, (float*)d_out);
}
